// EfficientPAM_51007031607666
// MI455X (gfx1250) — matrix operand order repaired and checked by reading, compiled, not run on hardware
//
#include <hip/hip_runtime.h>
#include <hip/hip_bf16.h>

typedef __attribute__((ext_vector_type(2)))  float  v2f;
typedef __attribute__((ext_vector_type(8)))  float  v8f;
typedef __attribute__((ext_vector_type(16))) __bf16 v16bf;
typedef int v4i_ __attribute__((vector_size(16)));
typedef __attribute__((address_space(1))) v4i_* as1_v4i;
typedef __attribute__((address_space(3))) v4i_* as3_v4i;

#define CH 64
#define DQ 8
#define HH 128
#define WW 128
#define NN (HH*WW)   // 16384
#define MM (NN/4)    // 4096
#define BB 4
#define NT (MM/32)   // 128 M-tiles of 32

#if __has_builtin(__builtin_amdgcn_global_load_async_to_lds_b128)
#define HAS_ASYNC 1
#else
#define HAS_ASYNC 0
#endif

static __device__ __forceinline__ unsigned short f2bf(float f) {
  unsigned u = __float_as_uint(f);
  u += 0x7fffu + ((u >> 16) & 1u);
  return (unsigned short)(u >> 16);
}

// pack two f32 -> packed bf16 pair (lo|hi<<16): 2 adds + 1 v_perm_b32
static __device__ __forceinline__ unsigned pkbf(float lo, float hi) {
  unsigned a = __float_as_uint(lo) + 0x7fffu;
  unsigned b = __float_as_uint(hi) + 0x7fffu;
  return __builtin_amdgcn_perm(b, a, 0x07060302); // {b[3],b[2],a[3],a[2]}
}

union BfPack { v16bf v; uint4 q[2]; };

// ---------------- Q = (Wq . x) / sqrt(d), stored [b*N + n][8] ----------------
__global__ void __launch_bounds__(256) qprep(const float* __restrict__ x,
                                             const float* __restrict__ Wq,
                                             float* __restrict__ Qws) {
  __shared__ float wq[DQ*CH];
  int tid = threadIdx.x;
  for (int i = tid; i < DQ*CH; i += 256) wq[i] = Wq[i];
  __syncthreads();
  int idx = blockIdx.x * 256 + tid;         // b*N + n
  float q[DQ];
#pragma unroll
  for (int d = 0; d < DQ; ++d) q[d] = 0.f;
  int b = idx >> 14, n = idx & (NN - 1);
  const float* xb = x + ((size_t)b * CH) * NN + n;
  for (int c = 0; c < CH; ++c) {
    float xv = xb[(size_t)c * NN];
#pragma unroll
    for (int d = 0; d < DQ; ++d) q[d] = fmaf(wq[d*CH + c], xv, q[d]);
  }
  const float scale = 0.3535533905932738f;  // 1/sqrt(8)
  float* qo = Qws + (size_t)idx * DQ;
#pragma unroll
  for (int d = 0; d < DQ; ++d) qo[d] = q[d] * scale;
}

// --- pooled x -> K^T [b*M + m][8] f32 ; V^T tiles [b][mt][c][mi] bf16 -------
__global__ void __launch_bounds__(256) kvprep(const float* __restrict__ x,
                                              const float* __restrict__ Wk,
                                              const float* __restrict__ Wv,
                                              float* __restrict__ Kws,
                                              unsigned short* __restrict__ Vws) {
  __shared__ float wk[DQ*CH];
  __shared__ float wv[CH*CH];
  int tid = threadIdx.x;
  for (int i = tid; i < DQ*CH; i += 256) wk[i] = Wk[i];
  for (int i = tid; i < CH*CH; i += 256) wv[i] = Wv[i];
  __syncthreads();
  int idx = blockIdx.x * 256 + tid;         // b*M + m
  int b = idx >> 12, m = idx & (MM - 1);
  int hp = m >> 6, wp = m & 63;
  float ka[DQ];
  float va[CH];
#pragma unroll
  for (int d = 0; d < DQ; ++d) ka[d] = 0.f;
#pragma unroll
  for (int c = 0; c < CH; ++c) va[c] = 0.f;
  const float* xb = x + ((size_t)b * CH) * NN + (2*hp)*WW + 2*wp;
  for (int c = 0; c < CH; ++c) {
    const float* p4 = xb + (size_t)c * NN;
    float p = 0.25f * (p4[0] + p4[1] + p4[WW] + p4[WW+1]);
#pragma unroll
    for (int d = 0; d < DQ; ++d) ka[d] = fmaf(wk[d*CH + c], p, ka[d]);
#pragma unroll
    for (int co = 0; co < CH; ++co) va[co] = fmaf(wv[co*CH + c], p, va[co]);
  }
  float* ko = Kws + (size_t)idx * DQ;
#pragma unroll
  for (int d = 0; d < DQ; ++d) ko[d] = ka[d];
  // V^T, tile-contiguous: Vws[((b*NT + m/32)*CH + c)*32 + m%32]
  unsigned short* vt = Vws + (((size_t)b * NT + (m >> 5)) * CH) * 32 + (m & 31);
#pragma unroll
  for (int c = 0; c < CH; ++c) vt[c * 32] = f2bf(va[c]);
}

// ------------------------- flash attention + epilogue ------------------------
__global__ void __launch_bounds__(256) flashattn(const float* __restrict__ x,
                                                 const float* __restrict__ gam,
                                                 const float* __restrict__ Qws,
                                                 const float* __restrict__ Kws,
                                                 const unsigned short* __restrict__ Vws,
                                                 float* __restrict__ out) {
  __shared__ __align__(16) unsigned short Vt[2][CH * 32]; // double-buffered V^T tile (bf16)

  const int tid  = threadIdx.x;
  const int wv   = tid >> 5;
  const int lane = tid & 31;
  const int half = lane >> 4;
  const int l16  = lane & 15;
  const int b    = blockIdx.y;
  const int n0   = blockIdx.x * 128 + wv * 16;

  // Q^T as f32-WMMA B operand (4x16): lane col n=l16; lanes0-15 k={k0,k0+1}, lanes16-31 k={k0+2,k0+3}
  const float* qb = Qws + ((size_t)b * NN + n0 + l16) * DQ + 2 * half;
  v2f bq0 = *reinterpret_cast<const v2f*>(qb);      // k 0..3 chunk
  v2f bq1 = *reinterpret_cast<const v2f*>(qb + 4);  // k 4..7 chunk

  v8f acc[4];
#pragma unroll
  for (int t = 0; t < 4; ++t)
#pragma unroll
    for (int v = 0; v < 8; ++v) acc[t][v] = 0.f;
  float rm = -1e30f, ls = 0.f;

  const float* kbase = Kws + (size_t)b * MM * DQ;
  const unsigned short* vtiles = Vws + ((size_t)b * NT) * (CH * 32);

  auto stage = [&](int tIdx, int buf) {
    const unsigned short* src = vtiles + (size_t)tIdx * (CH * 32) + tid * 8;
    unsigned short* dst = &Vt[buf][tid * 8];
#if HAS_ASYNC
    __builtin_amdgcn_global_load_async_to_lds_b128((as1_v4i)src, (as3_v4i)dst, 0, 0);
#else
    *reinterpret_cast<uint4*>(dst) = *reinterpret_cast<const uint4*>(src);
#endif
  };

  stage(0, 0);
  for (int it = 0; it < NT; ++it) {
    const int cur = it & 1;
#if HAS_ASYNC
    if (it + 1 < NT) { stage(it + 1, cur ^ 1);
                       asm volatile("s_wait_asynccnt 0x1" ::: "memory"); }
    else             { asm volatile("s_wait_asynccnt 0x0" ::: "memory"); }
#else
    if (it + 1 < NT) stage(it + 1, cur ^ 1);
#endif
    __syncthreads();   // tile `it` resident for all waves

    // ---- S^T = K^T @ Q^T : two 16(m)x16(n) f32 tiles (chained 16x16x4) ----
    v8f s[2];
#pragma unroll
    for (int sub = 0; sub < 2; ++sub) {
      // A 16x4: lane row m = l16; lanes0-15 k={k0,k0+1}, lanes16-31 k={k0+2,k0+3}
      const float* ka = kbase + (size_t)(it * 32 + sub * 16 + l16) * DQ + 2 * half;
      v2f a0 = *reinterpret_cast<const v2f*>(ka);
      v2f a1 = *reinterpret_cast<const v2f*>(ka + 4);
      v8f z = {0.f,0.f,0.f,0.f,0.f,0.f,0.f,0.f};
      z = __builtin_amdgcn_wmma_f32_16x16x4_f32(false, a0, false, bq0, (short)0, z, false, false);
      z = __builtin_amdgcn_wmma_f32_16x16x4_f32(false, a1, false, bq1, (short)0, z, false, false);
      s[sub] = z;  // lane(l16,h): s[sub][v] = S^T[m = sub*16 + v + 8h][n0 + l16]
    }

    // ---- online softmax over m: in-lane tree + one xor-16 exchange ----
    float mx;
    {
      v8f m8;
#pragma unroll
      for (int v = 0; v < 8; ++v) m8[v] = fmaxf(s[0][v], s[1][v]);
      float a = fmaxf(fmaxf(m8[0], m8[1]), fmaxf(m8[2], m8[3]));
      float c = fmaxf(fmaxf(m8[4], m8[5]), fmaxf(m8[6], m8[7]));
      mx = fmaxf(a, c);
      mx = fmaxf(mx, __shfl_xor(mx, 16, 32));
    }
    float nm   = fmaxf(rm, mx);
    float corr = __expf(rm - nm);
    float p0[8], p1[8];
#pragma unroll
    for (int v = 0; v < 8; ++v) { p0[v] = __expf(s[0][v] - nm); p1[v] = __expf(s[1][v] - nm); }
    float ps = 0.f;
#pragma unroll
    for (int v = 0; v < 8; ++v) ps += p0[v] + p1[v];
    ps += __shfl_xor(ps, 16, 32);
    ls = ls * corr + ps;
    rm = nm;
#pragma unroll
    for (int t = 0; t < 4; ++t)
#pragma unroll
      for (int v = 0; v < 8; ++v) acc[t][v] *= corr;

    // ---- build bf16 B operand P^T (32m x 16n) in registers ----
    // lanes0-15 need m=0..15: own p0 (m=v) + partner p0 (m=8+v)
    // lanes16-31 need m=16..31: partner p1 (m=16+v) + own p1 (m=24+v)
    unsigned pk0[4], pk1[4];
#pragma unroll
    for (int j = 0; j < 4; ++j) {
      pk0[j] = pkbf(p0[2*j], p0[2*j+1]);
      pk1[j] = pkbf(p1[2*j], p1[2*j+1]);
    }
    unsigned bu[8];
#pragma unroll
    for (int j = 0; j < 4; ++j) {
      unsigned send = half ? pk0[j] : pk1[j];
      unsigned recv = __shfl_xor(send, 16, 32);
      bu[j]     = half ? recv   : pk0[j];
      bu[j + 4] = half ? pk1[j] : recv;
    }
    BfPack pb;
    pb.q[0] = make_uint4(bu[0], bu[1], bu[2], bu[3]);
    pb.q[1] = make_uint4(bu[4], bu[5], bu[6], bu[7]);

    // ---- out^T += V^T @ P^T : four 16(c)x16(n) tiles via bf16 WMMA ----
#pragma unroll
    for (int t = 0; t < 4; ++t) {
      // A 16x32: lane row c = t*16 + l16; lanes0-15 K m={0..15}, lanes16-31 m={16..31}
      BfPack pa;
      const unsigned short* vr = &Vt[cur][(t * 16 + l16) * 32 + 16 * half];
      pa.q[0] = *reinterpret_cast<const uint4*>(vr);
      pa.q[1] = *reinterpret_cast<const uint4*>(vr + 8);
      acc[t] = __builtin_amdgcn_wmma_f32_16x16x32_bf16(false, pa.v, false, pb.v,
                                                       (short)0, acc[t], false, false);
    }
    __syncthreads();   // all reads of Vt[cur] done before it is re-staged
  }

  // ---- epilogue: y = x + gamma * acc / l ; acc[t][v]: c = t*16+v+8h, n = n0+l16 ----
  float g   = gam[0];
  float inv = 1.0f / ls;
#pragma unroll
  for (int t = 0; t < 4; ++t) {
#pragma unroll
    for (int v = 0; v < 8; ++v) {
      int c = t * 16 + v + 8 * half;
      size_t idx = ((size_t)b * CH + c) * NN + n0 + l16;
      out[idx] = x[idx] + g * acc[t][v] * inv;
    }
  }
}

extern "C" void kernel_launch(void* const* d_in, const int* in_sizes, int n_in,
                              void* d_out, int out_size, void* d_ws, size_t ws_size,
                              hipStream_t stream) {
  const float* x  = (const float*)d_in[0];
  const float* Wq = (const float*)d_in[1];
  const float* Wk = (const float*)d_in[2];
  const float* Wv = (const float*)d_in[3];
  const float* gm = (const float*)d_in[4];

  float* Qws = (float*)d_ws;                              // B*N*8 f32  (2 MB)
  float* Kws = Qws + (size_t)BB * NN * DQ;                // B*M*8 f32  (0.5 MB)
  unsigned short* Vws = (unsigned short*)(Kws + (size_t)BB * MM * DQ); // B*M*64 bf16 (2 MB)

  qprep <<<BB * NN / 256, 256, 0, stream>>>(x, Wq, Qws);
  kvprep<<<BB * MM / 256, 256, 0, stream>>>(x, Wk, Wv, Kws, Vws);
  dim3 grid(NN / 128, BB);
  flashattn<<<grid, 256, 0, stream>>>(x, gm, Qws, Kws, Vws, (float*)d_out);
}
